// SimpleGRUBackbone_71562745086174
// MI455X (gfx1250) — matrix operand order repaired and checked by reading, not compiled, not run on hardware
//
#include <hip/hip_runtime.h>
#include <hip/hip_bf16.h>

// ---------------------------------------------------------------------------
// SimpleGRUBackbone on MI455X (gfx1250, wave32, WMMA + TDM)
//   B=64, S=2048, D=256, V=256, L=4
//
// Per layer:
//   gi = seq_bf16 @ W_ih^T + b_ih      (bulk WMMA GEMM, fp32 out)
//   scan over t: gh = h_bf16 @ W_hh^T + b_hh (WMMA, W_hh register-resident),
//                gi tile (16x768 fp32, strided rows) double-buffered into LDS
//                by the Tensor Data Mover, overlapped with the WMMAs.
// Final: logits = seq_bf16 @ W_out^T + b_out (bulk WMMA GEMM)
// ---------------------------------------------------------------------------

typedef __bf16 bf16_t;
typedef unsigned int u32;
typedef __attribute__((ext_vector_type(16))) __bf16 v16bf;
typedef __attribute__((ext_vector_type(8)))  __bf16 v8bf;
typedef __attribute__((ext_vector_type(8)))  float  v8f;
typedef __attribute__((ext_vector_type(4)))  u32    v4u;
typedef __attribute__((ext_vector_type(8)))  int    v8i;
typedef __attribute__((ext_vector_type(4)))  int    v4i;

#define GRU_B 64
#define GRU_S 2048
#define GRU_D 256
#define GRU_V 256
#define GRU_L 4
#define GRU_G (3 * GRU_D)   // 768 gate columns

// ---- WMMA fragment builders (wave32; contiguous k order, same in A and B) ----------
// A: 16x32 bf16, lane holds row M=lane&15; element e -> K=hi*16+e
__device__ __forceinline__ v16bf load_a_frag(const bf16_t* __restrict__ Arow_k0,
                                             int hi) {
  return *(const v16bf*)(Arow_k0 + hi * 16);
}

// B: 32x16 bf16 (KxN), lane holds column N=lane&15; element e -> K=hi*16+e.
// Weights stored row-major [N,K] -> one contiguous 32-byte read per lane.
__device__ __forceinline__ v16bf load_b_frag(const bf16_t* __restrict__ Wrow,
                                             int k0, int hi) {
  return *(const v16bf*)(Wrow + k0 + hi * 16);
}

// ---- TDM: async 2D tile load (16 strided rows x 768 fp32) into LDS ---------
// Descriptor per CDNA5 ISA ch.8 (group0 + group1; 2D -> groups 2/3 zero).
// This toolchain exposes the 6-arg builtin:
//   (uint32x4 g0, int32x8 g1, int32x4, int32x4, int32x8, i32 cpol)
__device__ __forceinline__ void tdm_load_gi_tile(const float* gptr,
                                                 u32 lds_byte_off) {
  union { v4u v; u32 a[4]; } g0;
  union { v8i v; int a[8]; } g1;
  const unsigned long long ga = (unsigned long long)(uintptr_t)gptr;
  g0.a[0] = 1u;                                   // count=1 (valid user D#)
  g0.a[1] = lds_byte_off;                         // lds_addr
  g0.a[2] = (u32)ga;                              // global_addr[31:0]
  g0.a[3] = (u32)((ga >> 32) & 0x01FFFFFFu)       // global_addr[56:32]
            | (2u << 30);                         // type = 2 ("image")
  const u32 dim0 = GRU_G;                         // tensor_dim0 (elements)
  const u32 dim1 = (u32)GRU_B * (u32)GRU_S;       // tensor_dim1 (rows)
  const unsigned long long stride0 =
      (unsigned long long)GRU_S * (unsigned long long)GRU_G;  // row stride
  g1.a[0] = (int)(2u << 16);                      // data_size = 2 -> 4 bytes
  g1.a[1] = (int)((dim0 & 0xFFFFu) << 16);        // tensor_dim0[15:0]
  g1.a[2] = (int)(((dim0 >> 16) & 0xFFFFu) |      // tensor_dim0[31:16]
                  ((dim1 & 0xFFFFu) << 16));      // tensor_dim1[15:0]
  g1.a[3] = (int)(((dim1 >> 16) & 0xFFFFu) |      // tensor_dim1[31:16]
                  ((u32)GRU_G << 16));            // tile_dim0 = 768
  g1.a[4] = 16;                                   // tile_dim1 = 16 rows
  g1.a[5] = (int)(u32)(stride0 & 0xFFFFFFFFu);    // tensor_dim0_stride[31:0]
  g1.a[6] = (int)(u32)((stride0 >> 32) & 0xFFFFu);// tensor_dim0_stride[47:32]
  g1.a[7] = 0;
  const v4i z4 = {0, 0, 0, 0};
  const v8i z8 = {0, 0, 0, 0, 0, 0, 0, 0};
  __builtin_amdgcn_tensor_load_to_lds(g0.v, g1.v, z4, z4, z8, 0);
}

// ---- small helpers ---------------------------------------------------------
__global__ __launch_bounds__(256) void cast_f32_to_bf16(
    const float* __restrict__ in, bf16_t* __restrict__ out, int n) {
  int i = blockIdx.x * 256 + threadIdx.x;
  if (i < n) out[i] = (bf16_t)in[i];
}

__global__ __launch_bounds__(256) void embed_gather_bf16(
    const int* __restrict__ x, const float* __restrict__ emb,
    bf16_t* __restrict__ seq, int total) {
  int i = blockIdx.x * 256 + threadIdx.x;
  if (i < total) {
    int d  = i & (GRU_D - 1);
    int bs = i >> 8;  // D == 256
    seq[i] = (bf16_t)emb[(size_t)x[bs] * GRU_D + d];
  }
}

// ---- bulk GEMM: C[M,N] = A_bf16[M,K] @ W_bf16[N,K]^T + bias[N] -------------
__global__ __launch_bounds__(256) void gemm_bf16_wmma(
    const bf16_t* __restrict__ A, const bf16_t* __restrict__ W,
    const float* __restrict__ bias, float* __restrict__ C,
    int M, int N, int K) {
  const int wave = threadIdx.x >> 5;
  const int lane = threadIdx.x & 31;
  const int hi   = lane >> 4;
  const int nn   = lane & 15;
  const int m0   = (blockIdx.x * 8 + wave) * 16;
  const int n0   = blockIdx.y * 64;
  if (m0 >= M) return;

  v8f acc[4];
#pragma unroll
  for (int j = 0; j < 4; ++j) {
    const float bv = bias ? bias[n0 + j * 16 + nn] : 0.f;
#pragma unroll
    for (int e = 0; e < 8; ++e) acc[j][e] = bv;
  }

  const bf16_t* Arow = A + (size_t)(m0 + nn) * K;
  for (int k0 = 0; k0 < K; k0 += 32) {
    const v16bf a = load_a_frag(Arow + k0, hi);
#pragma unroll
    for (int j = 0; j < 4; ++j) {
      const bf16_t* Wrow = W + (size_t)(n0 + j * 16 + nn) * K;
      const v16bf b = load_b_frag(Wrow, k0, hi);
      acc[j] = __builtin_amdgcn_wmma_f32_16x16x32_bf16(
          false, a, false, b, (short)0, acc[j], false, false);
    }
  }

#pragma unroll
  for (int j = 0; j < 4; ++j)
#pragma unroll
    for (int r = 0; r < 8; ++r)
      C[(size_t)(m0 + r + 8 * hi) * N + n0 + j * 16 + nn] = acc[j][r];
}

// ---- persistent GRU scan with TDM-double-buffered gi -----------------------
// One block per 16-batch tile; h lives in LDS; W_hh fragments in VGPRs.
// Dynamic LDS layout:
//   [0,      16384)  h_f   : fp32 h[16][256]
//   [16384,  24576)  h_bf  : bf16 h[16][256]
//   [24576, 122880)  gi_l  : fp32 gi tile double buffer [2][16][768]
#define SCAN_LDS_HF   0
#define SCAN_LDS_HBF  16384
#define SCAN_LDS_GI   24576
#define SCAN_LDS_SIZE (24576 + 2 * 16 * GRU_G * 4)

__global__ __launch_bounds__(256, 2) void gru_scan_wmma(
    const float* __restrict__ gi,    // [B,S,3D] fp32 (precomputed input gates)
    const bf16_t* __restrict__ Wh,   // [3D,D] bf16, this layer
    const float* __restrict__ bh,    // [3D]
    bf16_t* __restrict__ seq_out) {  // [B,S,D] bf16 layer output
  extern __shared__ char smem[];
  float*  h_f  = (float*)(smem + SCAN_LDS_HF);
  bf16_t* h_bf = (bf16_t*)(smem + SCAN_LDS_HBF);
  float*  gi_l = (float*)(smem + SCAN_LDS_GI);
  const u32 lds_gi_base = (u32)(uintptr_t)(smem + SCAN_LDS_GI);

  const int wave = threadIdx.x >> 5;
  const int lane = threadIdx.x & 31;
  const int hi   = lane >> 4;
  const int nn   = lane & 15;
  const int b0   = blockIdx.x * 16;
  const int colbase = wave * 32;

  for (int i = threadIdx.x; i < 16 * GRU_D; i += 256) {
    h_f[i]  = 0.f;
    h_bf[i] = (bf16_t)0.f;
  }

  // Preload this wave's W_hh fragments + biases (time-invariant).
  v16bf bw[6][8];
  float bias[6];
#pragma unroll
  for (int gj = 0; gj < 6; ++gj) {
    const int g = gj >> 1, j = gj & 1;
    const int nrow = g * GRU_D + colbase + j * 16 + nn;
    bias[gj] = bh[nrow];
    const bf16_t* Wrow = Wh + (size_t)nrow * GRU_D;
#pragma unroll
    for (int ks = 0; ks < 8; ++ks) bw[gj][ks] = load_b_frag(Wrow, ks * 32, hi);
  }

  // Kick off the t=0 gi tile DMA (rows b0..b0+15 at timestep 0).
  if (wave == 0) {
    tdm_load_gi_tile(gi + (size_t)b0 * GRU_S * GRU_G, lds_gi_base);
  }
  __syncthreads();

  const bf16_t* Arow = h_bf + nn * GRU_D;  // lane's A row in LDS

  for (int t = 0; t < GRU_S; ++t) {
    // Issue DMA for next timestep's gi tile into the other buffer.
    if (t + 1 < GRU_S) {
      if (wave == 0) {
        tdm_load_gi_tile(
            gi + ((size_t)b0 * GRU_S + (size_t)(t + 1)) * GRU_G,
            lds_gi_base + (u32)(((t + 1) & 1) * 16 * GRU_G * 4));
      }
    }

    // gh = h @ Wh^T + bh for this wave's 6 gate tiles (overlaps with DMA)
    v8f acc[6];
#pragma unroll
    for (int gj = 0; gj < 6; ++gj)
#pragma unroll
      for (int e = 0; e < 8; ++e) acc[gj][e] = bias[gj];

#pragma unroll
    for (int ks = 0; ks < 8; ++ks) {
      const v16bf a = load_a_frag(Arow + ks * 32, hi);  // ds_load from LDS
#pragma unroll
      for (int gj = 0; gj < 6; ++gj)
        acc[gj] = __builtin_amdgcn_wmma_f32_16x16x32_bf16(
            false, a, false, bw[gj][ks], (short)0, acc[gj], false, false);
    }

    // Wave 0: ensure step-t tile has landed (<=1 leaves t+1's DMA in flight).
    if (t + 1 < GRU_S) {
      __builtin_amdgcn_s_wait_tensorcnt((short)1);
    } else {
      __builtin_amdgcn_s_wait_tensorcnt((short)0);
    }
    __syncthreads();  // gi tile visible to all waves; h reads complete

    // GRU gate math (fp32) + h update, register-local per wave
    const float* grow0 = gi_l + (size_t)(t & 1) * 16 * GRU_G;
#pragma unroll
    for (int j = 0; j < 2; ++j) {
      const int col = colbase + j * 16 + nn;
#pragma unroll
      for (int r = 0; r < 8; ++r) {
        const int m = r + 8 * hi;
        const float* grow = grow0 + m * GRU_G;
        const float ir  = grow[col];
        const float iz  = grow[GRU_D + col];
        const float in_ = grow[2 * GRU_D + col];
        const float rg = 1.f / (1.f + __expf(-(ir + acc[0 + j][r])));
        const float zg = 1.f / (1.f + __expf(-(iz + acc[2 + j][r])));
        const float ng = tanhf(in_ + rg * acc[4 + j][r]);
        const float hold = h_f[m * GRU_D + col];
        const float hnew = (1.f - zg) * ng + zg * hold;
        h_f [m * GRU_D + col] = hnew;
        h_bf[m * GRU_D + col] = (bf16_t)hnew;
        seq_out[((size_t)(b0 + m) * GRU_S + t) * GRU_D + col] = (bf16_t)hnew;
      }
    }
    __syncthreads();  // h fully updated before next step's WMMA reads
  }
}

// ---------------------------------------------------------------------------
extern "C" void kernel_launch(void* const* d_in, const int* in_sizes, int n_in,
                              void* d_out, int out_size, void* d_ws,
                              size_t ws_size, hipStream_t stream) {
  const int*   x     = (const int*)d_in[0];
  const float* emb   = (const float*)d_in[1];
  const float* W_ih  = (const float*)d_in[2];
  const float* W_hh  = (const float*)d_in[3];
  const float* b_ih  = (const float*)d_in[4];
  const float* b_hh  = (const float*)d_in[5];
  const float* W_out = (const float*)d_in[6];
  const float* b_out = (const float*)d_in[7];
  float* out = (float*)d_out;

  char* ws = (char*)d_ws;
  size_t off = 0;
  auto alloc = [&](size_t bytes) -> void* {
    void* p = ws + off;
    off = (off + bytes + 255) & ~(size_t)255;
    return p;
  };

  bf16_t* seqA  = (bf16_t*)alloc(sizeof(bf16_t) * GRU_B * GRU_S * GRU_D);
  bf16_t* seqB  = (bf16_t*)alloc(sizeof(bf16_t) * GRU_B * GRU_S * GRU_D);
  float*  gi    = (float*) alloc(sizeof(float)  * (size_t)GRU_B * GRU_S * GRU_G);
  bf16_t* Wi_bf = (bf16_t*)alloc(sizeof(bf16_t) * GRU_L * GRU_G * GRU_D);
  bf16_t* Wh_bf = (bf16_t*)alloc(sizeof(bf16_t) * GRU_L * GRU_G * GRU_D);
  bf16_t* Wo_bf = (bf16_t*)alloc(sizeof(bf16_t) * GRU_V * GRU_D);

  // Cast weights fp32 -> bf16
  const int nW = GRU_L * GRU_G * GRU_D;
  cast_f32_to_bf16<<<(nW + 255) / 256, 256, 0, stream>>>(W_ih, Wi_bf, nW);
  cast_f32_to_bf16<<<(nW + 255) / 256, 256, 0, stream>>>(W_hh, Wh_bf, nW);
  const int nWo = GRU_V * GRU_D;
  cast_f32_to_bf16<<<(nWo + 255) / 256, 256, 0, stream>>>(W_out, Wo_bf, nWo);

  // Embedding gather -> bf16 seq
  const int total = GRU_B * GRU_S * GRU_D;
  embed_gather_bf16<<<(total + 255) / 256, 256, 0, stream>>>(x, emb, seqA, total);

  const int Mrows = GRU_B * GRU_S;           // 131072
  dim3 gIn(Mrows / (16 * 8), GRU_G / 64);    // 1024 x 12
  dim3 gScan(GRU_B / 16);                    // 4 persistent blocks

  bf16_t* cur = seqA;
  bf16_t* nxt = seqB;
  for (int l = 0; l < GRU_L; ++l) {
    gemm_bf16_wmma<<<gIn, 256, 0, stream>>>(
        cur, Wi_bf + (size_t)l * GRU_G * GRU_D, b_ih + l * GRU_G, gi,
        Mrows, GRU_G, GRU_D);
    gru_scan_wmma<<<gScan, 256, SCAN_LDS_SIZE, stream>>>(
        gi, Wh_bf + (size_t)l * GRU_G * GRU_D, b_hh + l * GRU_G, nxt);
    bf16_t* tmp = cur; cur = nxt; nxt = tmp;
  }

  // logits = seq @ W_out^T + b_out -> fp32 d_out [B,S,V]
  dim3 gOut(Mrows / (16 * 8), GRU_V / 64);   // 1024 x 4
  gemm_bf16_wmma<<<gOut, 256, 0, stream>>>(cur, Wo_bf, b_out, out,
                                           Mrows, GRU_V, GRU_D);
}
